// ATR_17454747090989
// MI455X (gfx1250) — compile-verified, matrix operand order checked by reading, not run on hardware
//
#include <hip/hip_runtime.h>

// ---------------------------------------------------------------------------
// ATR twin-gate recurrence for MI455X (gfx1250, wave32, WMMA).
// Phase 1: p = x @ W1^T + b1  (32768x1024x1024 GEMM, bf16 WMMA, f32 accum)
// Phase 2: 256 sequential steps: q = h @ W2^T + b2; twin sigmoid gates.
// Pipeline: double-buffered LDS; B tile staged with CDNA5 async
// global->LDS copies (ASYNCcnt); A tile loads issued before the WMMAs and
// converted fp32->bf16 (hardware packed cvt) after them.
// ---------------------------------------------------------------------------

typedef __bf16 v16bf  __attribute__((ext_vector_type(16)));
typedef __bf16 bf16x4 __attribute__((ext_vector_type(4)));
typedef unsigned short u16x4  __attribute__((ext_vector_type(4)));
typedef unsigned short u16x8  __attribute__((ext_vector_type(8)));
typedef unsigned short u16x16 __attribute__((ext_vector_type(16)));
typedef float v8f __attribute__((ext_vector_type(8)));
typedef float v4f __attribute__((ext_vector_type(4)));

#define DD   1024   // feature dim
#define BB   128    // batch
#define TT   256    // time steps
#define BM   128    // block tile M
#define BN   128    // block tile N
#define BK   32     // K step (one bf16 WMMA)
#define LDST 40     // padded LDS row stride in u16 elements (80B = 20 banks)

// hardware fp32 -> bf16 converts (RNE), packed where the ISA allows
__device__ __forceinline__ u16x4 f2bf4(v4f f) {
    return __builtin_bit_cast(u16x4, __builtin_convertvector(f, bf16x4));
}
__device__ __forceinline__ unsigned short f2bf(float f) {
    return __builtin_bit_cast(unsigned short, (__bf16)f);
}

// Two async 16B global->LDS copies per thread: one 32B B-tile chunk.
// GVS addressing: mem = SGPR64 + VGPR32 + IOFFSET; IOFFSET is added to BOTH
// the memory address and the LDS address, so offset:16 covers the 2nd chunk.
__device__ __forceinline__ void async_copy_b32B(unsigned lds_addr,
                                                unsigned voff,
                                                unsigned long long base)
{
    asm volatile(
        "global_load_async_to_lds_b128 %0, %1, %2\n\t"
        "global_load_async_to_lds_b128 %0, %1, %2 offset:16"
        :: "v"(lds_addr), "v"(voff), "s"(base) : "memory");
}

__device__ __forceinline__ void wait_async0() {
    asm volatile("s_wait_asynccnt 0x0" ::: "memory");
}

// Core tiled GEMM: C(128x128) += A(f32, row-major, ld=DD) * Bw(bf16 as u16,
// stored [n][k] row-major, ld=DD)^T   i.e. C[m][n] = sum_k A[m][k]*Bw[n][k].
__device__ __forceinline__ void gemm_core(const float* __restrict__ A,
                                          const unsigned short* __restrict__ Bw,
                                          int mBase, int nBase,
                                          v8f (&c)[2][4])
{
    __shared__ unsigned short As[2][BM * LDST];
    __shared__ unsigned short Bs[2][BN * LDST];

    const int tid  = threadIdx.x;
    const int wave = tid >> 5;          // 0..7
    const int lane = tid & 31;
    const int wm   = wave >> 1;         // 0..3  (wave row)
    const int wn   = wave & 1;          // 0..1  (wave col)
    const int lh   = lane & 15;
    const int ls   = lane >> 4;         // half-wave select
    const int kseg = ls * 8;            // K sub-segment per ISA A/B layout
    const int lrow = tid >> 1;          // 0..127 loader row
    const int lcol = (tid & 1) * 16;    // 0 or 16 loader col (k)

    // staging addresses for this thread
    const float* aSrc = A + (size_t)(mBase + lrow) * DD + lcol;
    const unsigned long long bBase =
        (unsigned long long)(Bw + (size_t)nBase * DD);
    const unsigned bVOff = (unsigned)(lrow * DD + lcol) * 2u;  // bytes

    // LDS byte addresses (low 32 bits of the flat pointer = LDS offset)
    const unsigned ldsB0 = (unsigned)(size_t)&Bs[0][lrow * LDST + lcol];
    const unsigned ldsB1 = (unsigned)(size_t)&Bs[1][lrow * LDST + lcol];

    // ---- prologue: stage tile 0 into buffer 0 ----
    {
        v4f f0 = *(const v4f*)(aSrc + 0);
        v4f f1 = *(const v4f*)(aSrc + 4);
        v4f f2 = *(const v4f*)(aSrc + 8);
        v4f f3 = *(const v4f*)(aSrc + 12);
        async_copy_b32B(ldsB0, bVOff, bBase);
        u16x4 c0 = f2bf4(f0), c1 = f2bf4(f1), c2 = f2bf4(f2), c3 = f2bf4(f3);
        u16x8 a0, a1;
        #pragma unroll
        for (int i = 0; i < 4; ++i) {
            a0[i] = c0[i]; a0[4 + i] = c1[i];
            a1[i] = c2[i]; a1[4 + i] = c3[i];
        }
        *(u16x8*)&As[0][lrow * LDST + lcol]     = a0;
        *(u16x8*)&As[0][lrow * LDST + lcol + 8] = a1;
        wait_async0();
    }
    __syncthreads();

    #pragma unroll 2
    for (int k0 = 0; k0 < DD; k0 += BK) {
        const int  cur     = (k0 >> 5) & 1;
        const int  nxt     = cur ^ 1;
        const bool hasNext = (k0 + BK) < DD;

        // ---- issue next tile's loads early (overlap with WMMA) ----
        v4f f0, f1, f2, f3;
        if (hasNext) {
            const float* sa = aSrc + (k0 + BK);
            f0 = *(const v4f*)(sa + 0);
            f1 = *(const v4f*)(sa + 4);
            f2 = *(const v4f*)(sa + 8);
            f3 = *(const v4f*)(sa + 12);
            async_copy_b32B(nxt ? ldsB1 : ldsB0,
                            bVOff + (unsigned)(k0 + BK) * 2u, bBase);
        }

        // ---- fragment loads (ISA 16-bit A 16x32 layout) ----
        v16bf afr[2], bfr[4];
        #pragma unroll
        for (int tm = 0; tm < 2; ++tm) {
            int r = wm * 32 + tm * 16 + lh;
            u16x8 lo = *(const u16x8*)&As[cur][r * LDST + kseg];
            u16x8 hi = *(const u16x8*)&As[cur][r * LDST + 16 + kseg];
            u16x16 v;
            #pragma unroll
            for (int i = 0; i < 8; ++i) { v[i] = lo[i]; v[8 + i] = hi[i]; }
            afr[tm] = __builtin_bit_cast(v16bf, v);
        }
        #pragma unroll
        for (int tn = 0; tn < 4; ++tn) {
            int r = wn * 64 + tn * 16 + lh;
            u16x8 lo = *(const u16x8*)&Bs[cur][r * LDST + kseg];
            u16x8 hi = *(const u16x8*)&Bs[cur][r * LDST + 16 + kseg];
            u16x16 v;
            #pragma unroll
            for (int i = 0; i < 8; ++i) { v[i] = lo[i]; v[8 + i] = hi[i]; }
            bfr[tn] = __builtin_bit_cast(v16bf, v);
        }

        // ---- 8 WMMAs per wave per K step ----
        #pragma unroll
        for (int tm = 0; tm < 2; ++tm)
            #pragma unroll
            for (int tn = 0; tn < 4; ++tn)
                c[tm][tn] = __builtin_amdgcn_wmma_f32_16x16x32_bf16(
                    false, afr[tm], false, bfr[tn],
                    (short)0, c[tm][tn], false, false);

        // ---- finish staging next tile after the math ----
        if (hasNext) {
            u16x4 c0 = f2bf4(f0), c1 = f2bf4(f1), c2 = f2bf4(f2), c3 = f2bf4(f3);
            u16x8 a0, a1;
            #pragma unroll
            for (int i = 0; i < 4; ++i) {
                a0[i] = c0[i]; a0[4 + i] = c1[i];
                a1[i] = c2[i]; a1[4 + i] = c3[i];
            }
            *(u16x8*)&As[nxt][lrow * LDST + lcol]     = a0;
            *(u16x8*)&As[nxt][lrow * LDST + lcol + 8] = a1;
            wait_async0();
        }
        __syncthreads();
    }
}

// ---- Phase 1: p[m][n] = sum_k x[m][k] * W1[n][k] + b1[n] --------------------
__global__ void __launch_bounds__(256)
atr_gemm_p(const float* __restrict__ x, const unsigned short* __restrict__ W1bf,
           const float* __restrict__ b1, float* __restrict__ p)
{
    const int nBase = blockIdx.x * BN;
    const int mBase = blockIdx.y * BM;
    v8f c[2][4] = {};
    gemm_core(x, W1bf, mBase, nBase, c);

    const int tid = threadIdx.x;
    const int wave = tid >> 5, lane = tid & 31;
    const int wm = wave >> 1, wn = wave & 1;
    const int lh = lane & 15, ls = lane >> 4;

    #pragma unroll
    for (int tm = 0; tm < 2; ++tm)
        #pragma unroll
        for (int tn = 0; tn < 4; ++tn) {
            int n = nBase + wn * 64 + tn * 16 + lh;
            float bias = b1[n];
            #pragma unroll
            for (int i = 0; i < 8; ++i) {
                int m = mBase + wm * 32 + tm * 16 + ls * 8 + i;
                p[(size_t)m * DD + n] = c[tm][tn][i] + bias;
            }
        }
}

// ---- Phase 2: one recurrence step ------------------------------------------
// q = hin @ W2^T + b2 ; ig = sig(p+q) ; fg = sig(p-q) ; h = ig*p + fg*q
__global__ void __launch_bounds__(256)
atr_step(const float* __restrict__ hin, float* __restrict__ hout,
         const float* __restrict__ p, const unsigned short* __restrict__ W2bf,
         const float* __restrict__ b2, float* __restrict__ out, int t)
{
    const int nBase = blockIdx.x * BN;
    v8f c[2][4] = {};
    gemm_core(hin, W2bf, 0, nBase, c);

    const int tid = threadIdx.x;
    const int wave = tid >> 5, lane = tid & 31;
    const int wm = wave >> 1, wn = wave & 1;
    const int lh = lane & 15, ls = lane >> 4;

    #pragma unroll
    for (int tm = 0; tm < 2; ++tm)
        #pragma unroll
        for (int tn = 0; tn < 4; ++tn) {
            int n = nBase + wn * 64 + tn * 16 + lh;
            float bias = b2[n];
            #pragma unroll
            for (int i = 0; i < 8; ++i) {
                int m = wm * 32 + tm * 16 + ls * 8 + i;   // batch index
                float q  = c[tm][tn][i] + bias;
                size_t po = (size_t)m * (TT * DD) + (size_t)t * DD + n;
                float pt = p[po];
                float ig = 1.0f / (1.0f + __expf(-(pt + q)));
                float fg = 1.0f / (1.0f + __expf(-(pt - q)));
                float hn = ig * pt + fg * q;
                hout[(size_t)m * DD + n] = hn;
                out[po] = hn;
            }
        }
}

// ---- helpers ----------------------------------------------------------------
__global__ void atr_cvt_weights(const float* __restrict__ W1,
                                const float* __restrict__ W2,
                                unsigned short* __restrict__ W1bf,
                                unsigned short* __restrict__ W2bf, int n)
{
    int i = blockIdx.x * blockDim.x + threadIdx.x;
    if (i < n) {
        W1bf[i] = f2bf(W1[i]);
        W2bf[i] = f2bf(W2[i]);
    }
}

__global__ void atr_init_h(const float* __restrict__ init_hx,
                           float* __restrict__ h)
{
    int i = blockIdx.x * blockDim.x + threadIdx.x;
    h[i] = init_hx[i & (DD - 1)];
}

// ---------------------------------------------------------------------------
extern "C" void kernel_launch(void* const* d_in, const int* in_sizes, int n_in,
                              void* d_out, int out_size, void* d_ws, size_t ws_size,
                              hipStream_t stream)
{
    const float* x       = (const float*)d_in[0];   // [B,T,D]
    const float* W1      = (const float*)d_in[1];   // [D,D]
    const float* b1      = (const float*)d_in[2];   // [D]
    const float* W2      = (const float*)d_in[3];   // [D,D]
    const float* b2      = (const float*)d_in[4];   // [D]
    const float* init_hx = (const float*)d_in[5];   // [1,D]
    float* out = (float*)d_out;                      // [B,T,D]

    char* ws = (char*)d_ws;
    float* p = (float*)ws;                                    // 128 MB fp32
    size_t pBytes = (size_t)BB * TT * DD * sizeof(float);
    unsigned short* W1bf = (unsigned short*)(ws + pBytes);    // 2 MB
    unsigned short* W2bf = W1bf + (size_t)DD * DD;            // 2 MB
    float* h0 = (float*)(W2bf + (size_t)DD * DD);             // 512 KB
    float* h1 = h0 + (size_t)BB * DD;                         // 512 KB

    // 1) convert weights to bf16 (L2-resident for the whole scan)
    atr_cvt_weights<<<(DD * DD + 255) / 256, 256, 0, stream>>>(W1, W2, W1bf, W2bf, DD * DD);

    // 2) h0 = broadcast(init_hx)
    atr_init_h<<<(BB * DD) / 256, 256, 0, stream>>>(init_hx, h0);

    // 3) big input GEMM: p = x @ W1^T + b1
    atr_gemm_p<<<dim3(DD / BN, (BB * TT) / BM), 256, 0, stream>>>(x, W1bf, b1, p);

    // 4) sequential twin-gate scan (256 dependent launches, ping-pong h)
    float* hcur = h0;
    float* hnxt = h1;
    for (int t = 0; t < TT; ++t) {
        atr_step<<<dim3(DD / BN, 1), 256, 0, stream>>>(hcur, hnxt, p, W2bf, b2, out, t);
        float* tmp = hcur; hcur = hnxt; hnxt = tmp;
    }
}
